// LayoutGnn_27943057228427
// MI455X (gfx1250) — compile-verified, matrix operand order checked by reading, not run on hardware
//
#include <hip/hip_runtime.h>
#include <math.h>

// ---------------------------------------------------------------------------
// LayoutGnn for MI455X (gfx1250): 5x GATv2 + residual + graph-LayerNorm.
// Dense transforms via v_wmma_f32_16x16x32_bf16 (fp32 accumulate); each wave
// computes a 16x(NT*16) output strip with NT a template constant (no branches
// in the k-loop -> no spills, EXEC all-ones). A fragment loaded once per
// k-step feeds NT WMMAs. Edge softmax/aggregation in fp32 with L2 float
// atomics. Working set (~45 MB) fits in the 192 MB L2.
// ---------------------------------------------------------------------------

#define QN    27
#define FEAT  32
#define BGR   256
#define NN    (BGR * QN)     // 6912 nodes
#define DD    (3 * FEAT)     // 96
#define HHID  (DD * 4)       // 384
#define CPH   DD             // channels per head (96 for every layer)
#define EMAX  82944
#define EPMAX (EMAX + NN)    // edges incl. self loops = 89856
#define LN_EPS 1e-5f

typedef __bf16 bf16_t;
typedef bf16_t v16bf __attribute__((ext_vector_type(16)));
typedef float  v8f   __attribute__((ext_vector_type(8)));
typedef unsigned int u32x4 __attribute__((ext_vector_type(4)));

__device__ __forceinline__ unsigned short f2bf(float f) {
    unsigned u = __float_as_uint(f);
    u += 0x7FFFu + ((u >> 16) & 1u);          // round-to-nearest-even
    return (unsigned short)(u >> 16);
}

__device__ __forceinline__ void atomicMaxF(float* addr, float val) {
    unsigned* ua = (unsigned*)addr;
    unsigned old = *ua;
    while (true) {
        float f = __uint_as_float(old);
        if (f >= val) break;
        unsigned assumed = old;
        old = atomicCAS(ua, assumed, __float_as_uint(val));
        if (old == assumed) break;
    }
}

// ------------------------- setup kernels -----------------------------------

__global__ void k_fill(float* p, float v, int n) {
    int i = blockIdx.x * blockDim.x + threadIdx.x;
    if (i < n) p[i] = v;
}

// h0 = [x | repeat(circ_feat, Q)] ; keep fp32 residual and bf16 GEMM input
__global__ void k_concat(const float* __restrict__ x, const float* __restrict__ circ,
                         float* __restrict__ resid, unsigned short* __restrict__ hbf) {
    int i = blockIdx.x * blockDim.x + threadIdx.x;
    if (i >= NN * DD) return;
    int node = i / DD, c = i - node * DD;
    float v = (c < FEAT) ? x[node * FEAT + c]
                         : circ[(node / QN) * (2 * FEAT) + (c - FEAT)];
    resid[i] = v;
    hbf[i] = f2bf(v);
}

// W[K x M] fp32  ->  WT[M x K] bf16 (so B fragments are K-contiguous)
__global__ void k_wprep(const float* __restrict__ W, unsigned short* __restrict__ WT,
                        int K, int M) {
    int i = blockIdx.x * blockDim.x + threadIdx.x;
    if (i >= K * M) return;
    int k = i / M, m = i - k * M;
    WT[(size_t)m * K + k] = f2bf(W[i]);
}

// ------------------------- WMMA GEMM ---------------------------------------
// C[N x M] = A[N x K](bf16) * B[K x M], B given as BT[M x K](bf16).
// One wave per 16x(NT*16) strip, NT compile-time; grids are sized so every
// strip is full -> zero guards, EXEC stays all-ones (WMMA requirement).
// Fragment layout (wave32): lane%16 = row(A)/col(B), lane/16 selects K half;
// vector elems {0..7}=K[half*8..+7], {8..15}=K[16+half*8..].
template <int NT>
__global__ void k_gemm_bf16(const unsigned short* __restrict__ A,
                            const unsigned short* __restrict__ BT,
                            float* __restrict__ C, int K, int M) {
    const int strip = blockIdx.y * blockDim.y + threadIdx.y;
    const int lane  = threadIdx.x;
    const int l15   = lane & 15;
    const int hf    = lane >> 4;

    const unsigned short* arow = A  + (size_t)(blockIdx.x * 16 + l15) * K;
    const unsigned short* brow = BT + (size_t)(strip * NT * 16 + l15) * K;

    v8f acc[NT];
#pragma unroll
    for (int j = 0; j < NT; ++j) acc[j] = (v8f){};

    for (int k0 = 0; k0 < K; k0 += 32) {
        union { v16bf v; u32x4 u[2]; } fa;
        fa.u[0] = *(const u32x4*)(arow + k0 +      hf * 8);
        fa.u[1] = *(const u32x4*)(arow + k0 + 16 + hf * 8);
#pragma unroll
        for (int j = 0; j < NT; ++j) {
            union { v16bf v; u32x4 u[2]; } fb;
            const unsigned short* bj = brow + (size_t)j * 16 * K;
            fb.u[0] = *(const u32x4*)(bj + k0 +      hf * 8);
            fb.u[1] = *(const u32x4*)(bj + k0 + 16 + hf * 8);
            acc[j] = __builtin_amdgcn_wmma_f32_16x16x32_bf16(
                         false, fa.v, false, fb.v, (short)0, acc[j], false, false);
        }
    }
    // D layout: lane%16 = column, (lane/16)*8 + r = row
#pragma unroll
    for (int j = 0; j < NT; ++j) {
        float* cbase = C + (size_t)(blockIdx.x * 16) * M + (strip * NT + j) * 16;
#pragma unroll
        for (int r = 0; r < 8; ++r)
            cbase[(size_t)(hf * 8 + r) * M + l15] = acc[j][r];
    }
}

// ------------------------- edge pipeline -----------------------------------

// wave per (edge, head): logits = sum_c lrelu(xl[src]+xr[dst],0.2)*att ;
// segment max into Mx[dst,h] via CAS float-max.
__global__ void k_logits(const long long* __restrict__ ei,
                         const float* __restrict__ XL, const float* __restrict__ XR,
                         const float* __restrict__ att,
                         float* __restrict__ logits, float* __restrict__ Mx,
                         int E, int Ep, int Hh, int OUT) {
    int wave = blockIdx.x * (blockDim.x >> 5) + (threadIdx.x >> 5);
    int lane = threadIdx.x & 31;
    if (wave >= Ep * Hh) return;
    int e = wave / Hh, h = wave - e * Hh;
    int src, dst;
    if (e < E) { src = (int)ei[e]; dst = (int)ei[E + e]; }
    else       { src = dst = e - E; }

    const float* xl = XL + (size_t)src * OUT + h * CPH;
    const float* xr = XR + (size_t)dst * OUT + h * CPH;
    const float* at = att + h * CPH;
    float s = 0.f;
#pragma unroll
    for (int c = lane; c < CPH; c += 32) {
        float v = xl[c] + xr[c];
        v = v > 0.f ? v : 0.2f * v;
        s += v * at[c];
    }
#pragma unroll
    for (int m = 16; m >= 1; m >>= 1) s += __shfl_xor(s, m, 32);
    if (lane == 0) {
        logits[wave] = s;
        atomicMaxF(&Mx[(size_t)dst * Hh + h], s);
    }
}

// a = exp(logit - max); denom[dst,h] += a
__global__ void k_expnorm(const long long* __restrict__ ei,
                          float* __restrict__ logits, const float* __restrict__ Mx,
                          float* __restrict__ denom, int E, int Ep, int Hh) {
    int t = blockIdx.x * blockDim.x + threadIdx.x;
    if (t >= Ep * Hh) return;
    int e = t / Hh, h = t - e * Hh;
    int dst = (e < E) ? (int)ei[E + e] : (e - E);
    float a = __expf(logits[t] - Mx[(size_t)dst * Hh + h]);
    logits[t] = a;
    atomicAdd(&denom[(size_t)dst * Hh + h], a);
}

// out[dst] += alpha * xl[src]   (wave per (edge, head), 3 atomics/lane)
__global__ void k_agg(const long long* __restrict__ ei,
                      const float* __restrict__ logits, const float* __restrict__ denom,
                      const float* __restrict__ XL, float* __restrict__ acc,
                      int E, int Ep, int Hh, int OUT) {
    int wave = blockIdx.x * (blockDim.x >> 5) + (threadIdx.x >> 5);
    int lane = threadIdx.x & 31;
    if (wave >= Ep * Hh) return;
    int e = wave / Hh, h = wave - e * Hh;
    int src, dst;
    if (e < E) { src = (int)ei[e]; dst = (int)ei[E + e]; }
    else       { src = dst = e - E; }
    float alpha = logits[wave] / denom[(size_t)dst * Hh + h];
    const float* xl = XL + (size_t)src * OUT + h * CPH;
    float* ap = acc + (size_t)dst * OUT + h * CPH;
#pragma unroll
    for (int c = lane; c < CPH; c += 32)
        atomicAdd(&ap[c], alpha * xl[c]);
}

// inner layers: h = lrelu(acc + b, 0.01) -> bf16 input for next GEMM
__global__ void k_finalize(const float* __restrict__ acc, const float* __restrict__ bias,
                           unsigned short* __restrict__ hbf, int OUT) {
    int i = blockIdx.x * blockDim.x + threadIdx.x;
    if (i >= NN * OUT) return;
    int c = i % OUT;
    float v = acc[i] + bias[c];
    v = v > 0.f ? v : 0.01f * v;
    hbf[i] = f2bf(v);
}

// last layer: h = lrelu(acc + b, 0.01) + residual
__global__ void k_finalize_last(const float* __restrict__ acc, const float* __restrict__ bias,
                                const float* __restrict__ resid, float* __restrict__ hout) {
    int i = blockIdx.x * blockDim.x + threadIdx.x;
    if (i >= NN * DD) return;
    int c = i % DD;
    float v = acc[i] + bias[c];
    v = v > 0.f ? v : 0.01f * v;
    hout[i] = v + resid[i];
}

// ------------------------- graph LayerNorm ---------------------------------

__global__ void k_lnreduce(const float* __restrict__ h, float* __restrict__ sums) {
    __shared__ float s0[256], s1[256];
    float a = 0.f, b = 0.f;
    for (int i = blockIdx.x * blockDim.x + threadIdx.x; i < NN * DD;
         i += gridDim.x * blockDim.x) {
        float v = h[i];
        a += v; b += v * v;
    }
    s0[threadIdx.x] = a; s1[threadIdx.x] = b;
    __syncthreads();
    for (int st = 128; st > 0; st >>= 1) {
        if (threadIdx.x < st) {
            s0[threadIdx.x] += s0[threadIdx.x + st];
            s1[threadIdx.x] += s1[threadIdx.x + st];
        }
        __syncthreads();
    }
    if (threadIdx.x == 0) {
        atomicAdd(&sums[0], s0[0]);
        atomicAdd(&sums[1], s1[0]);
    }
}

__global__ void k_lnapply(const float* __restrict__ h, const float* __restrict__ sums,
                          const float* __restrict__ w, const float* __restrict__ b,
                          float* __restrict__ out) {
    int i = blockIdx.x * blockDim.x + threadIdx.x;
    if (i >= NN * DD) return;
    const float invNT = 1.f / (float)(NN * DD);
    float mu  = sums[0] * invNT;
    float var = sums[1] * invNT - mu * mu;
    float inv = rsqrtf(var + LN_EPS);
    int c = i % DD;
    out[i] = (h[i] - mu) * inv * w[c] + b[c];
}

// ------------------------- host orchestration -------------------------------

extern "C" void kernel_launch(void* const* d_in, const int* in_sizes, int n_in,
                              void* d_out, int out_size, void* d_ws, size_t ws_size,
                              hipStream_t stream) {
    (void)n_in; (void)out_size; (void)ws_size;
    const float*      circ   = (const float*)d_in[0];
    const float*      x      = (const float*)d_in[1];
    const long long*  ei     = (const long long*)d_in[2];
    const float*      WlF    = (const float*)d_in[3];
    const float*      WrF    = (const float*)d_in[4];
    const float*      attF   = (const float*)d_in[5];
    const float*      bF     = (const float*)d_in[6];
    const float*      WlI    = (const float*)d_in[7];
    const float*      WrI    = (const float*)d_in[8];
    const float*      attI   = (const float*)d_in[9];
    const float*      bI     = (const float*)d_in[10];
    const float*      WlL    = (const float*)d_in[11];
    const float*      WrL    = (const float*)d_in[12];
    const float*      attL   = (const float*)d_in[13];
    const float*      bL     = (const float*)d_in[14];
    const float*      lnw    = (const float*)d_in[15];
    const float*      lnb    = (const float*)d_in[16];

    const int E  = in_sizes[2] / 2;   // 82944
    const int Ep = E + NN;

    // --- workspace carve-out (256B aligned) ---
    char* base = (char*)d_ws;
    size_t off = 0;
    auto carve = [&](size_t bytes) -> char* {
        char* p = base + off;
        off += (bytes + 255) & ~(size_t)255;
        return p;
    };
    unsigned short* HBF   = (unsigned short*)carve((size_t)NN * HHID * 2);
    float*          XL    = (float*)carve((size_t)NN * HHID * 4);
    float*          XR    = (float*)carve((size_t)NN * HHID * 4);
    float*          ACC   = (float*)carve((size_t)NN * HHID * 4);
    float*          RESID = (float*)carve((size_t)NN * DD * 4);
    float*          HLAST = (float*)carve((size_t)NN * DD * 4);
    float*          LOGB  = (float*)carve((size_t)EPMAX * 4 * 4);
    float*          MX    = (float*)carve((size_t)NN * 4 * 4);
    float*          DEN   = (float*)carve((size_t)NN * 4 * 4);
    unsigned short* WTP   = (unsigned short*)carve((size_t)1032192 * 2);
    float*          SUMS  = (float*)carve(256);

    // transposed bf16 weights inside WTP
    unsigned short* WlFt = WTP;                        // 384*96
    unsigned short* WrFt = WlFt + 36864;
    unsigned short* WlIt = WrFt + 36864;               // 3 * 384*384
    unsigned short* WrIt = WlIt + 3 * 147456;
    unsigned short* WlLt = WrIt + 3 * 147456;          // 96*384
    unsigned short* WrLt = WlLt + 36864;

    const int TPB = 256;
    auto blocks = [](long long n, int t) { return (int)((n + t - 1) / t); };

    // 0) build h0 (residual fp32 + bf16)
    k_concat<<<blocks((long long)NN * DD, TPB), TPB, 0, stream>>>(x, circ, RESID, HBF);

    // 1) weight prep: fp32 [K x M] -> bf16 [M x K]
    k_wprep<<<blocks(96 * 384, TPB), TPB, 0, stream>>>(WlF, WlFt, 96, 384);
    k_wprep<<<blocks(96 * 384, TPB), TPB, 0, stream>>>(WrF, WrFt, 96, 384);
    for (int i = 0; i < 3; ++i) {
        k_wprep<<<blocks(384 * 384, TPB), TPB, 0, stream>>>(WlI + (size_t)i * 147456,
                                                            WlIt + (size_t)i * 147456, 384, 384);
        k_wprep<<<blocks(384 * 384, TPB), TPB, 0, stream>>>(WrI + (size_t)i * 147456,
                                                            WrIt + (size_t)i * 147456, 384, 384);
    }
    k_wprep<<<blocks(384 * 96, TPB), TPB, 0, stream>>>(WlL, WlLt, 384, 96);
    k_wprep<<<blocks(384 * 96, TPB), TPB, 0, stream>>>(WrL, WrLt, 384, 96);

    // 2) five GATv2 layers
    struct Layer {
        int K, H, OUT;
        const unsigned short *wlT, *wrT;
        const float *att, *bias;
    } L[5] = {
        {  96, 4, HHID, WlFt, WrFt, attF, bF },
        { HHID, 4, HHID, WlIt + 0 * 147456, WrIt + 0 * 147456, attI + 0 * 384, bI + 0 * 384 },
        { HHID, 4, HHID, WlIt + 1 * 147456, WrIt + 1 * 147456, attI + 1 * 384, bI + 1 * 384 },
        { HHID, 4, HHID, WlIt + 2 * 147456, WrIt + 2 * 147456, attI + 2 * 384, bI + 2 * 384 },
        { HHID, 1,   DD, WlLt, WrLt, attL, bL },
    };

    for (int li = 0; li < 5; ++li) {
        const Layer& ly = L[li];
        const int tasks = Ep * ly.H;

        // GEMMs: XL = h @ Wl, XR = h @ Wr   (v_wmma_f32_16x16x32_bf16)
        // OUT=384: 24 tiles = 6 strips of NT=4 (grid.y=3, 2 waves/block)
        // OUT=96 :  6 tiles = 2 strips of NT=3 (grid.y=1, 2 waves/block)
        dim3 gB(32, 2);
        if (ly.OUT == HHID) {
            dim3 gG(NN / 16, 3);
            k_gemm_bf16<4><<<gG, gB, 0, stream>>>(HBF, ly.wlT, XL, ly.K, ly.OUT);
            k_gemm_bf16<4><<<gG, gB, 0, stream>>>(HBF, ly.wrT, XR, ly.K, ly.OUT);
        } else {
            dim3 gG(NN / 16, 1);
            k_gemm_bf16<3><<<gG, gB, 0, stream>>>(HBF, ly.wlT, XL, ly.K, ly.OUT);
            k_gemm_bf16<3><<<gG, gB, 0, stream>>>(HBF, ly.wrT, XR, ly.K, ly.OUT);
        }

        // reset scratch
        k_fill<<<blocks((long long)NN * ly.OUT, TPB), TPB, 0, stream>>>(ACC, 0.f, NN * ly.OUT);
        k_fill<<<blocks((long long)NN * ly.H, TPB), TPB, 0, stream>>>(DEN, 0.f, NN * ly.H);
        k_fill<<<blocks((long long)NN * ly.H, TPB), TPB, 0, stream>>>(MX, -INFINITY, NN * ly.H);

        // attention
        k_logits<<<blocks(tasks, 8), 256, 0, stream>>>(ei, XL, XR, ly.att, LOGB, MX,
                                                       E, Ep, ly.H, ly.OUT);
        k_expnorm<<<blocks(tasks, TPB), TPB, 0, stream>>>(ei, LOGB, MX, DEN, E, Ep, ly.H);
        k_agg<<<blocks(tasks, 8), 256, 0, stream>>>(ei, LOGB, DEN, XL, ACC,
                                                    E, Ep, ly.H, ly.OUT);

        if (li < 4) {
            k_finalize<<<blocks((long long)NN * ly.OUT, TPB), TPB, 0, stream>>>(
                ACC, ly.bias, HBF, ly.OUT);
        } else {
            k_finalize_last<<<blocks((long long)NN * DD, TPB), TPB, 0, stream>>>(
                ACC, ly.bias, RESID, HLAST);
        }
    }

    // 3) graph LayerNorm over all N*D elements -> d_out
    k_fill<<<1, 64, 0, stream>>>(SUMS, 0.f, 2);
    k_lnreduce<<<256, 256, 0, stream>>>(HLAST, SUMS);
    k_lnapply<<<blocks((long long)NN * DD, TPB), TPB, 0, stream>>>(
        HLAST, SUMS, lnw, lnb, (float*)d_out);
}
